// EdgeDecoder_47090021433540
// MI455X (gfx1250) — matrix operand order repaired and checked by reading, not compiled, not run on hardware
//
#include <hip/hip_runtime.h>
#include <hip/hip_bf16.h>

typedef __attribute__((ext_vector_type(16))) _Float16 v16h;
typedef __attribute__((ext_vector_type(8)))  float    v8f;

#define WAVES_PER_BLOCK 8
#define BLOCK_THREADS   (WAVES_PER_BLOCK * 32)
// W1 fragment store: 8 k-tiles * 4 n-tiles * 32 lanes * 16 halves = 16384 f16 = 32 KB
#define W1_FRAG_ELEMS   16384

// ---------------------------------------------------------------------------
// Prep: convert W1 [256][64] fp32 (row-major [k][n]) into f16 B-fragments.
// Fragment f = kt*4+nt. Per wave32 B layout (32x16, 16-bit):
//   lanes 0..15  : column n = nt*16 + lane,      elems j=0..7 -> K = kt*32 + j,     j=8..15 -> K = kt*32 + 16 + (j-8)
//   lanes 16..31 : column n = nt*16 + (lane-16), elems j=0..7 -> K = kt*32 + 8 + j, j=8..15 -> K = kt*32 + 24 + (j-8)
// Stored flat: ws[((f*32 + lane)*16) + j]
// ---------------------------------------------------------------------------
__global__ void prep_w1_frags(const float* __restrict__ w1,
                              _Float16* __restrict__ wsB) {
    int tid = blockIdx.x * blockDim.x + threadIdx.x;
    if (tid < W1_FRAG_ELEMS) {
        int j    = tid & 15;
        int lane = (tid >> 4) & 31;
        int f    = tid >> 9;          // 0..31
        int kt   = f >> 2;
        int nt   = f & 3;
        int n    = nt * 16 + (lane & 15);
        int k    = kt * 32 + ((lane >> 4) ? 8 : 0) + (j & 7) + ((j >> 3) * 16);
        wsB[tid] = (_Float16)w1[k * 64 + n];
    }
}

// ---------------------------------------------------------------------------
// Main: one wave = one 16-edge tile. A built by per-lane row gather + cvt;
// W1 B-fragments are staged in LDS once per block and end up hoisted into
// VGPRs by the compiler (loop-invariant), so the steady-state loop is just
// gather loads + packed cvts + 32 x v_wmma_f32_16x16x32_f16 per tile.
// Next-tile software prefetch gives a full tile of L2-gather lookahead.
// NOTE: no min-waves launch bound — capping VGPRs makes LLVM spill the
// hoisted fragments to scratch (seen in round 2), which is far worse.
// ---------------------------------------------------------------------------
__global__ void __launch_bounds__(BLOCK_THREADS)
edge_mlp_wmma(const float* __restrict__ emb,        // [N_NODES][128] fp32
              const long long* __restrict__ ei,     // [2][E] int64
              const _Float16* __restrict__ wsB,     // W1 fragments (32 KB)
              const float* __restrict__ b1,         // [64]
              const float* __restrict__ w2,         // [64]
              const float* __restrict__ b2,         // [1]
              float* __restrict__ out,              // [E]
              int E, int ntiles) {
    __shared__ __align__(32) _Float16 ldsB[W1_FRAG_ELEMS];   // 32 KB

    // Cooperative copy of W1 fragments into LDS (2048 x 16B).
    {
        const int4* src = (const int4*)wsB;
        int4*       dst = (int4*)ldsB;
        for (int i = threadIdx.x; i < W1_FRAG_ELEMS / 8; i += BLOCK_THREADS)
            dst[i] = src[i];
    }
    __syncthreads();

    const int lane = threadIdx.x & 31;
    const int wave = threadIdx.x >> 5;
    const int m    = lane & 15;   // row within 16-edge tile
    const int hf   = lane >> 4;   // K-half select

    // Epilogue params: this lane's hidden column for each N-tile is nt*16+m.
    float b1v[4], w2v[4];
#pragma unroll
    for (int nt = 0; nt < 4; ++nt) {
        b1v[nt] = b1[nt * 16 + m];
        w2v[nt] = w2[nt * 16 + m];
    }
    const float b2s = b2[0];

    const int wave_stride = gridDim.x * WAVES_PER_BLOCK;
    for (int t = blockIdx.x * WAVES_PER_BLOCK + wave; t < ntiles; t += wave_stride) {
        int e = t * 16 + m;
        if (e >= E) e = E - 1;                 // clamp tail, keep EXEC full
        const long long s = ei[e];
        const long long d = ei[(long long)E + e];
        const float* rowS = emb + s * 128;
        const float* rowD = emb + d * 128;

        // ---- prefetch next tile's gathered rows (full-tile lookahead over
        //      the 32 WMMAs below). Each lane covers its K-half: bytes
        //      [hf*256, hf*256+256) of both 512B rows.
        {
            int tn = t + wave_stride;
            if (tn < ntiles) {                 // wave-uniform guard
                int e2 = tn * 16 + m;
                if (e2 >= E) e2 = E - 1;
                const long long s2 = ei[e2];
                const long long d2 = ei[(long long)E + e2];
                const float* p0 = emb + s2 * 128 + hf * 64;
                const float* p1 = emb + d2 * 128 + hf * 64;
                __builtin_prefetch(p0, 0, 3);
                __builtin_prefetch(p0 + 32, 0, 3);
                __builtin_prefetch(p1, 0, 3);
                __builtin_prefetch(p1 + 32, 0, 3);
            }
        }

        v8f acc0 = {}, acc1 = {}, acc2 = {}, acc3 = {};

#pragma unroll
        for (int kt = 0; kt < 8; ++kt) {
            // K-step kt covers f16-K [kt*32, kt*32+32); kt<4 -> src row, else dst.
            const float* row  = (kt < 4) ? rowS : rowD;
            const int    base = ((kt & 3) * 32) + hf * 8;
            // A layout (16-bit 16x32): lane holds K = base..base+7, base+16..base+23
            float4 f0 = *(const float4*)(row + base);
            float4 f1 = *(const float4*)(row + base + 4);
            float4 f2 = *(const float4*)(row + base + 16);
            float4 f3 = *(const float4*)(row + base + 20);
            v16h a;
            a[0]  = (_Float16)f0.x; a[1]  = (_Float16)f0.y;
            a[2]  = (_Float16)f0.z; a[3]  = (_Float16)f0.w;
            a[4]  = (_Float16)f1.x; a[5]  = (_Float16)f1.y;
            a[6]  = (_Float16)f1.z; a[7]  = (_Float16)f1.w;
            a[8]  = (_Float16)f2.x; a[9]  = (_Float16)f2.y;
            a[10] = (_Float16)f2.z; a[11] = (_Float16)f2.w;
            a[12] = (_Float16)f3.x; a[13] = (_Float16)f3.y;
            a[14] = (_Float16)f3.z; a[15] = (_Float16)f3.w;

            const _Float16* bbase = ldsB + (kt * 4) * 32 * 16 + lane * 16;
            v16h bf0 = *(const v16h*)(bbase + 0 * 32 * 16);
            v16h bf1 = *(const v16h*)(bbase + 1 * 32 * 16);
            v16h bf2 = *(const v16h*)(bbase + 2 * 32 * 16);
            v16h bf3 = *(const v16h*)(bbase + 3 * 32 * 16);

            acc0 = __builtin_amdgcn_wmma_f32_16x16x32_f16(false, a, false, bf0,
                                                          (short)0, acc0, false, false);
            acc1 = __builtin_amdgcn_wmma_f32_16x16x32_f16(false, a, false, bf1,
                                                          (short)0, acc1, false, false);
            acc2 = __builtin_amdgcn_wmma_f32_16x16x32_f16(false, a, false, bf2,
                                                          (short)0, acc2, false, false);
            acc3 = __builtin_amdgcn_wmma_f32_16x16x32_f16(false, a, false, bf3,
                                                          (short)0, acc3, false, false);
        }

        // Epilogue. C layout: VGPR r -> row (r + 8*hf), lane%16 -> column.
        // logit[row] = b2 + sum_cols relu(c + b1[col]) * w2[col]
#pragma unroll
        for (int r = 0; r < 8; ++r) {
            float s0 = acc0[r] + b1v[0]; s0 = s0 > 0.f ? s0 : 0.f;
            float s1 = acc1[r] + b1v[1]; s1 = s1 > 0.f ? s1 : 0.f;
            float s2 = acc2[r] + b1v[2]; s2 = s2 > 0.f ? s2 : 0.f;
            float s3 = acc3[r] + b1v[3]; s3 = s3 > 0.f ? s3 : 0.f;
            float v = fmaf(s0, w2v[0],
                      fmaf(s1, w2v[1],
                      fmaf(s2, w2v[2], s3 * w2v[3])));
            // reduce over the 16-lane column group (stays within each half)
            v += __shfl_xor(v, 1, 32);
            v += __shfl_xor(v, 2, 32);
            v += __shfl_xor(v, 4, 32);
            v += __shfl_xor(v, 8, 32);
            int orow = t * 16 + r + 8 * hf;
            if (m == 0 && orow < E)
                out[orow] = v + b2s;
        }
    }
}

// ---------------------------------------------------------------------------
extern "C" void kernel_launch(void* const* d_in, const int* in_sizes, int n_in,
                              void* d_out, int out_size, void* d_ws, size_t ws_size,
                              hipStream_t stream) {
    const float*     emb = (const float*)d_in[0];
    const long long* ei  = (const long long*)d_in[1];   // int64 per reference
    const float*     w1  = (const float*)d_in[2];
    const float*     b1  = (const float*)d_in[3];
    const float*     w2  = (const float*)d_in[4];
    const float*     b2  = (const float*)d_in[5];
    float*           out = (float*)d_out;
    _Float16*        wsB = (_Float16*)d_ws;

    const int E      = in_sizes[1] / 2;     // edge_index is [2, E]
    const int ntiles = (E + 15) / 16;

    prep_w1_frags<<<(W1_FRAG_ELEMS + 255) / 256, 256, 0, stream>>>(w1, wsB);

    int blocks = 1024;                      // grid-stride; ~4 tiles per wave
    int max_blocks = (ntiles + WAVES_PER_BLOCK - 1) / WAVES_PER_BLOCK;
    if (blocks > max_blocks) blocks = max_blocks;
    edge_mlp_wmma<<<blocks, BLOCK_THREADS, 0, stream>>>(emb, ei, wsB, b1, w2, b2,
                                                        out, E, ntiles);
}
